// RSSM_36764920054546
// MI455X (gfx1250) — matrix operand order repaired and checked by reading, not compiled, not run on hardware
//
#include <hip/hip_runtime.h>
#include <math.h>

// ---------------------------------------------------------------------------
// Types for CDNA5 WMMA (wave32): A/B = 16 bf16 per lane, C/D = 8 f32 per lane
// ---------------------------------------------------------------------------
typedef __attribute__((ext_vector_type(16))) __bf16 v16bf;
typedef __attribute__((ext_vector_type(8)))  float  v8f;

struct FragU {
    union { v16bf v; uint4 q[2]; };
};

// ---------------------------------------------------------------------------
// bf16 helpers: round-to-nearest-even conversion + hi/lo split (bf16x3 scheme)
// ---------------------------------------------------------------------------
__device__ __forceinline__ unsigned short f2bf_rn(float f) {
    unsigned u = __float_as_uint(f);
    u += 0x7FFFu + ((u >> 16) & 1u);
    return (unsigned short)(u >> 16);
}
__device__ __forceinline__ float bf2f(unsigned short h) {
    return __uint_as_float(((unsigned)h) << 16);
}
__device__ __forceinline__ void split_bf16(float x, unsigned short& hi, unsigned short& lo) {
    hi = f2bf_rn(x);
    lo = f2bf_rn(x - bf2f(hi));
}

// ---------------------------------------------------------------------------
// Fragment loaders.
// A (16x32 bf16, row-major source, leading dim = ld elements):
//   lane l: row = l&15, g = l>>4; holds K = [16g..16g+15]
//   -> two contiguous 16B runs -> two global_load_b128.
// B: weights pre-swizzled so lane l's 16 elements for tile ti are contiguous
//   (element e of lane l corresponds to K = 16*(l>>4)+e, N = l&15).
// ---------------------------------------------------------------------------
__device__ __forceinline__ void load_a_frag(FragU& f, const unsigned short* base,
                                            int row, int ld, int k0, int g) {
    const uint4* p = reinterpret_cast<const uint4*>(base + (size_t)row * ld + k0);
    f.q[0] = p[2 * g];
    f.q[1] = p[2 * g + 1];
}
__device__ __forceinline__ void load_b_frag(FragU& f, const unsigned short* bw,
                                            int ti, int lane) {
    const uint4* p = reinterpret_cast<const uint4*>(bw + (((size_t)ti * 32 + lane) << 4));
    f.q[0] = p[0];
    f.q[1] = p[1];
}

// 3-term bf16 emulation of an fp32 MAC tile: hi*hi + hi*lo + lo*hi (f32 accum)
__device__ __forceinline__ void mma3(v8f& c, const FragU& ah, const FragU& al,
                                     const FragU& bh, const FragU& bl) {
    c = __builtin_amdgcn_wmma_f32_16x16x32_bf16(false, ah.v, false, bh.v, (short)0, c, false, false);
    c = __builtin_amdgcn_wmma_f32_16x16x32_bf16(false, ah.v, false, bl.v, (short)0, c, false, false);
    c = __builtin_amdgcn_wmma_f32_16x16x32_bf16(false, al.v, false, bh.v, (short)0, c, false, false);
}

__device__ __forceinline__ float sigmoidf_(float x) { return 1.f / (1.f + __expf(-x)); }
__device__ __forceinline__ float eluf_(float x)     { return x > 0.f ? x : (__expf(x) - 1.f); }
__device__ __forceinline__ float softplusf_(float x){ return x > 20.f ? x : log1pf(__expf(x)); }

// ---------------------------------------------------------------------------
// Weight conversion: f32 [N][K] row-major (torch Linear layout; B(k,n)=W[n][k])
// -> swizzled bf16 hi/lo, fragment-ready. Tile ti = nt*KC + kc.
// ---------------------------------------------------------------------------
__global__ void convert_weight_kernel(const float* __restrict__ W,
                                      unsigned short* __restrict__ outH,
                                      unsigned short* __restrict__ outL,
                                      int K, int KC, int total) {
    int idx = blockIdx.x * blockDim.x + threadIdx.x;
    if (idx >= total) return;
    int ti   = idx >> 9;
    int rem  = idx & 511;
    int lane = rem >> 4;
    int e    = rem & 15;
    int ntt  = ti / KC;
    int kc   = ti - ntt * KC;
    int n = ntt * 16 + (lane & 15);
    int k = kc * 32 + (lane >> 4) * 16 + e;
    float w = W[(size_t)n * K + k];
    unsigned short hi, lo;
    split_bf16(w, hi, lo);
    outH[idx] = hi;
    outL[idx] = lo;
}

__global__ void init_state_kernel(float* __restrict__ hF,
                                  unsigned short* __restrict__ hH,
                                  unsigned short* __restrict__ hL,
                                  float* __restrict__ sF) {
    int idx = blockIdx.x * blockDim.x + threadIdx.x;
    if (idx < 256 * 1024) { hF[idx] = 0.f; hH[idx] = 0; hL[idx] = 0; }
    if (idx < 256 * 64) sF[idx] = 0.f;
}

// ---------------------------------------------------------------------------
// Merged per-step pack: [s|a] concat (256x96) and obs_{t+1} (256x1024) -> bf16 hi/lo
// ---------------------------------------------------------------------------
__global__ void pack_step_kernel(const float* __restrict__ sCur,
                                 const float* __restrict__ actAll,
                                 const float* __restrict__ obsAll, int t,
                                 unsigned short* __restrict__ saH, unsigned short* __restrict__ saL,
                                 unsigned short* __restrict__ obH, unsigned short* __restrict__ obL) {
    int idx = blockIdx.x * blockDim.x + threadIdx.x;
    const int NSA = 256 * 96;
    if (idx < NSA) {
        int b = idx / 96, c = idx - b * 96;
        float x = (c < 64) ? sCur[(size_t)b * 64 + c]
                           : actAll[((size_t)b * 64 + t) * 32 + (c - 64)];
        unsigned short hi, lo; split_bf16(x, hi, lo);
        saH[idx] = hi; saL[idx] = lo;
    } else {
        int j = idx - NSA;
        if (j >= 256 * 1024) return;
        int b = j >> 10, c = j & 1023;
        float x = obsAll[((size_t)b * 64 + (t + 1)) * 1024 + c];
        unsigned short hi, lo; split_bf16(x, hi, lo);
        obH[j] = hi; obL[j] = lo;
    }
}

// ---------------------------------------------------------------------------
// e = ELU([s|a] @ Wsa^T + bsa): M=256, N=1024, K=96 (KC=3).
// Block = 256 threads (8 waves); block tile 16(M) x 128(N); wave tile 16x16.
// ---------------------------------------------------------------------------
__global__ void __launch_bounds__(256)
sa_embed_kernel(const unsigned short* __restrict__ Ah, const unsigned short* __restrict__ Al,
                const unsigned short* __restrict__ Bh, const unsigned short* __restrict__ Bl,
                const float* __restrict__ bias,
                unsigned short* __restrict__ outH, unsigned short* __restrict__ outL) {
    const int lane = threadIdx.x & 31;
    const int wv   = threadIdx.x >> 5;
    const int g    = lane >> 4;
    const int nl   = lane & 15;
    const int m0   = blockIdx.x * 16;
    const int n0   = blockIdx.y * 128 + wv * 16;
    const int nt   = n0 >> 4;
    const int rowA = m0 + nl;
    const int KC   = 3;
    v8f c = {};
    #pragma unroll
    for (int kc = 0; kc < KC; ++kc) {
        FragU fah, fal, fbh, fbl;
        load_a_frag(fah, Ah, rowA, 96, kc * 32, g);
        load_a_frag(fal, Al, rowA, 96, kc * 32, g);
        load_b_frag(fbh, Bh, nt * KC + kc, lane);
        load_b_frag(fbl, Bl, nt * KC + kc, lane);
        mma3(c, fah, fal, fbh, fbl);
    }
    #pragma unroll
    for (int i = 0; i < 8; ++i) {
        int row = m0 + i + 8 * g;
        int col = n0 + nl;
        float y = eluf_(c[i] + bias[col]);
        unsigned short hi, lo; split_bf16(y, hi, lo);
        outH[(size_t)row * 1024 + col] = hi;
        outL[(size_t)row * 1024 + col] = lo;
    }
}

// ---------------------------------------------------------------------------
// Fused GRU step: gi = e@Wih^T, gh = h@Whh^T (6 accumulators: r/z/n x {i,h}),
// gates + h_new in epilogue. Writes h_new (f32 + bf16 hi/lo) and output slice.
// K = 1024 (KC=32); Wih/Whh have N=3072 -> gate col-tile offsets 0/64/128.
// ---------------------------------------------------------------------------
__global__ void __launch_bounds__(256)
gru_kernel(const unsigned short* __restrict__ eH, const unsigned short* __restrict__ eL,
           const unsigned short* __restrict__ hH, const unsigned short* __restrict__ hL,
           const float* __restrict__ hOld,
           const unsigned short* __restrict__ WihH, const unsigned short* __restrict__ WihL,
           const unsigned short* __restrict__ WhhH, const unsigned short* __restrict__ WhhL,
           const float* __restrict__ bih, const float* __restrict__ bhh,
           float* __restrict__ hNewF,
           unsigned short* __restrict__ hNewH, unsigned short* __restrict__ hNewL,
           float* __restrict__ outSlice) {
    const int lane = threadIdx.x & 31;
    const int wv   = threadIdx.x >> 5;
    const int g    = lane >> 4;
    const int nl   = lane & 15;
    const int m0   = blockIdx.x * 16;
    const int n0   = blockIdx.y * 128 + wv * 16;
    const int nt   = n0 >> 4;               // 0..63 inside each gate block
    const int KC   = 32;
    const int rowA = m0 + nl;
    v8f cir = {}, ciz = {}, cia = {}, chr_ = {}, chz = {}, cha = {};
    for (int kc = 0; kc < KC; ++kc) {
        FragU aeh, ael, ahh, ahl, bh, bl;
        load_a_frag(aeh, eH, rowA, 1024, kc * 32, g);
        load_a_frag(ael, eL, rowA, 1024, kc * 32, g);
        load_a_frag(ahh, hH, rowA, 1024, kc * 32, g);
        load_a_frag(ahl, hL, rowA, 1024, kc * 32, g);
        // input-side gates
        load_b_frag(bh, WihH, (nt      ) * KC + kc, lane);
        load_b_frag(bl, WihL, (nt      ) * KC + kc, lane);
        mma3(cir, aeh, ael, bh, bl);
        load_b_frag(bh, WihH, (64 + nt ) * KC + kc, lane);
        load_b_frag(bl, WihL, (64 + nt ) * KC + kc, lane);
        mma3(ciz, aeh, ael, bh, bl);
        load_b_frag(bh, WihH, (128 + nt) * KC + kc, lane);
        load_b_frag(bl, WihL, (128 + nt) * KC + kc, lane);
        mma3(cia, aeh, ael, bh, bl);
        // hidden-side gates
        load_b_frag(bh, WhhH, (nt      ) * KC + kc, lane);
        load_b_frag(bl, WhhL, (nt      ) * KC + kc, lane);
        mma3(chr_, ahh, ahl, bh, bl);
        load_b_frag(bh, WhhH, (64 + nt ) * KC + kc, lane);
        load_b_frag(bl, WhhL, (64 + nt ) * KC + kc, lane);
        mma3(chz, ahh, ahl, bh, bl);
        load_b_frag(bh, WhhH, (128 + nt) * KC + kc, lane);
        load_b_frag(bl, WhhL, (128 + nt) * KC + kc, lane);
        mma3(cha, ahh, ahl, bh, bl);
    }
    #pragma unroll
    for (int i = 0; i < 8; ++i) {
        int row = m0 + i + 8 * g;
        int col = n0 + nl;
        float xr = cir[i] + bih[col]        + chr_[i] + bhh[col];
        float xz = ciz[i] + bih[1024 + col] + chz[i]  + bhh[1024 + col];
        float r  = sigmoidf_(xr);
        float z  = sigmoidf_(xz);
        float xa = cia[i] + bih[2048 + col] + r * (cha[i] + bhh[2048 + col]);
        float n  = tanhf(xa);
        float ho = hOld[(size_t)row * 1024 + col];
        float hn = (1.f - z) * n + z * ho;
        hNewF[(size_t)row * 1024 + col] = hn;
        unsigned short hi, lo; split_bf16(hn, hi, lo);
        hNewH[(size_t)row * 1024 + col] = hi;
        hNewL[(size_t)row * 1024 + col] = lo;
        outSlice[(size_t)row * 1408 + col] = hn;
    }
}

// ---------------------------------------------------------------------------
// Merged hidden heads: blockIdx.z == 0 -> p1 = ELU(h_new @ Wp1^T + bp1), K=1024
//                      blockIdx.z == 1 -> q1 = ELU([h_new|obs] @ Wq1^T + bq1), K=2048
// Grid (16, 8, 2); block = 256 threads (8 waves); wave tile 16x16.
// ---------------------------------------------------------------------------
__global__ void __launch_bounds__(256)
heads_hidden_kernel(const unsigned short* __restrict__ hH, const unsigned short* __restrict__ hL,
                    const unsigned short* __restrict__ obH, const unsigned short* __restrict__ obL,
                    const unsigned short* __restrict__ Wp1H, const unsigned short* __restrict__ Wp1L,
                    const float* __restrict__ bp1,
                    const unsigned short* __restrict__ Wq1H, const unsigned short* __restrict__ Wq1L,
                    const float* __restrict__ bq1,
                    unsigned short* __restrict__ p1H, unsigned short* __restrict__ p1L,
                    unsigned short* __restrict__ q1H, unsigned short* __restrict__ q1L) {
    const int z    = blockIdx.z;
    const int lane = threadIdx.x & 31;
    const int wv   = threadIdx.x >> 5;
    const int g    = lane >> 4;
    const int nl   = lane & 15;
    const int m0   = blockIdx.x * 16;
    const int n0   = blockIdx.y * 128 + wv * 16;
    const int nt   = n0 >> 4;
    const int rowA = m0 + nl;
    const unsigned short* Bh   = z ? Wq1H : Wp1H;
    const unsigned short* Bl   = z ? Wq1L : Wp1L;
    const float*          bias = z ? bq1  : bp1;
    const int KC = z ? 64 : 32;
    v8f c = {};
    for (int kc = 0; kc < KC; ++kc) {
        const unsigned short* ah = (kc < 32) ? hH : obH;
        const unsigned short* al = (kc < 32) ? hL : obL;
        int k0 = (kc & 31) * 32;
        FragU fah, fal, fbh, fbl;
        load_a_frag(fah, ah, rowA, 1024, k0, g);
        load_a_frag(fal, al, rowA, 1024, k0, g);
        load_b_frag(fbh, Bh, nt * KC + kc, lane);
        load_b_frag(fbl, Bl, nt * KC + kc, lane);
        mma3(c, fah, fal, fbh, fbl);
    }
    unsigned short* outH = z ? q1H : p1H;
    unsigned short* outL = z ? q1L : p1L;
    #pragma unroll
    for (int i = 0; i < 8; ++i) {
        int row = m0 + i + 8 * g;
        int col = n0 + nl;
        float y = eluf_(c[i] + bias[col]);
        unsigned short hi, lo; split_bf16(y, hi, lo);
        outH[(size_t)row * 1024 + col] = hi;
        outL[(size_t)row * 1024 + col] = lo;
    }
}

// ---------------------------------------------------------------------------
// Merged output heads: blockIdx.y == 0 -> prior (p1 @ Wp2^T), base 1024
//                      blockIdx.y == 1 -> posterior (q1 @ Wq2^T), base 1216,
//                                         also writes next stoch state.
// N = 128: cols 0..63 mean, 64..127 raw std; fused softplus + reparam sample.
// Grid (16, 2); block = 128 threads (4 waves): wave w owns mean tile w, std 4+w.
// ---------------------------------------------------------------------------
__global__ void __launch_bounds__(128)
heads_out_kernel(const unsigned short* __restrict__ p1H, const unsigned short* __restrict__ p1L,
                 const unsigned short* __restrict__ q1H, const unsigned short* __restrict__ q1L,
                 const unsigned short* __restrict__ Wp2H, const unsigned short* __restrict__ Wp2L,
                 const float* __restrict__ bp2,
                 const unsigned short* __restrict__ Wq2H, const unsigned short* __restrict__ Wq2L,
                 const float* __restrict__ bq2,
                 const float* __restrict__ nzp, const float* __restrict__ nzq,
                 float* __restrict__ outSlice, float* __restrict__ sNext) {
    const int z    = blockIdx.y;
    const int lane = threadIdx.x & 31;
    const int wv   = threadIdx.x >> 5;   // 0..3
    const int g    = lane >> 4;
    const int nl   = lane & 15;
    const int m0   = blockIdx.x * 16;
    const int KC   = 32;
    const int rowA = m0 + nl;
    const unsigned short* Ah = z ? q1H : p1H;
    const unsigned short* Al = z ? q1L : p1L;
    const unsigned short* Bh = z ? Wq2H : Wp2H;
    const unsigned short* Bl = z ? Wq2L : Wp2L;
    const float* bias  = z ? bq2 : bp2;
    const float* noise = z ? nzq : nzp;
    const int outBase  = z ? 1216 : 1024;
    v8f cm = {}, cs = {};
    for (int kc = 0; kc < KC; ++kc) {
        FragU ah, al, bh, bl;
        load_a_frag(ah, Ah, rowA, 1024, kc * 32, g);
        load_a_frag(al, Al, rowA, 1024, kc * 32, g);
        load_b_frag(bh, Bh, (wv    ) * KC + kc, lane);
        load_b_frag(bl, Bl, (wv    ) * KC + kc, lane);
        mma3(cm, ah, al, bh, bl);
        load_b_frag(bh, Bh, (4 + wv) * KC + kc, lane);
        load_b_frag(bl, Bl, (4 + wv) * KC + kc, lane);
        mma3(cs, ah, al, bh, bl);
    }
    #pragma unroll
    for (int i = 0; i < 8; ++i) {
        int row = m0 + i + 8 * g;
        int col = wv * 16 + nl;                 // 0..63
        float mean = cm[i] + bias[col];
        float stdv = softplusf_(cs[i] + bias[64 + col]) + 0.1f;
        float st   = mean + stdv * noise[(size_t)row * 64 + col];
        float* orow = outSlice + (size_t)row * 1408;
        orow[outBase + col]       = mean;
        orow[outBase + 64 + col]  = stdv;
        orow[outBase + 128 + col] = st;
        if (z) sNext[(size_t)row * 64 + col] = st;
    }
}

// ---------------------------------------------------------------------------
// Host side
// ---------------------------------------------------------------------------
extern "C" void kernel_launch(void* const* d_in, const int* in_sizes, int n_in,
                              void* d_out, int out_size, void* d_ws, size_t ws_size,
                              hipStream_t stream) {
    (void)in_sizes; (void)n_in; (void)out_size; (void)ws_size;
    const float* obs = (const float*)d_in[0];
    const float* act = (const float*)d_in[1];
    const float* nzp = (const float*)d_in[2];
    const float* nzq = (const float*)d_in[3];
    const float* Wsa = (const float*)d_in[4];
    const float* bsa = (const float*)d_in[5];
    const float* Wih = (const float*)d_in[6];
    const float* bih = (const float*)d_in[7];
    const float* Whh = (const float*)d_in[8];
    const float* bhh = (const float*)d_in[9];
    const float* Wp1 = (const float*)d_in[10];
    const float* bp1 = (const float*)d_in[11];
    const float* Wp2 = (const float*)d_in[12];
    const float* bp2 = (const float*)d_in[13];
    const float* Wq1 = (const float*)d_in[14];
    const float* bq1 = (const float*)d_in[15];
    const float* Wq2 = (const float*)d_in[16];
    const float* bq2 = (const float*)d_in[17];
    float* out = (float*)d_out;

    // bump allocator over d_ws (256B aligned slabs)
    char* p = (char*)d_ws;
    auto alloc_us = [&](size_t elems) -> unsigned short* {
        unsigned short* r = (unsigned short*)p;
        p += (elems * 2 + 255) & ~(size_t)255;
        return r;
    };
    auto alloc_f = [&](size_t elems) -> float* {
        float* r = (float*)p;
        p += (elems * 4 + 255) & ~(size_t)255;
        return r;
    };

    // swizzled bf16 hi/lo weights
    unsigned short *WsaH = alloc_us(1024 * 96),   *WsaL = alloc_us(1024 * 96);
    unsigned short *WihH = alloc_us(3072 * 1024), *WihL = alloc_us(3072 * 1024);
    unsigned short *WhhH = alloc_us(3072 * 1024), *WhhL = alloc_us(3072 * 1024);
    unsigned short *Wp1H = alloc_us(1024 * 1024), *Wp1L = alloc_us(1024 * 1024);
    unsigned short *Wp2H = alloc_us(128 * 1024),  *Wp2L = alloc_us(128 * 1024);
    unsigned short *Wq1H = alloc_us(1024 * 2048), *Wq1L = alloc_us(1024 * 2048);
    unsigned short *Wq2H = alloc_us(128 * 1024),  *Wq2L = alloc_us(128 * 1024);
    // activations
    unsigned short *saH = alloc_us(256 * 96),   *saL = alloc_us(256 * 96);
    unsigned short *eH  = alloc_us(256 * 1024), *eL  = alloc_us(256 * 1024);
    unsigned short *obH = alloc_us(256 * 1024), *obL = alloc_us(256 * 1024);
    unsigned short *p1H = alloc_us(256 * 1024), *p1L = alloc_us(256 * 1024);
    unsigned short *q1H = alloc_us(256 * 1024), *q1L = alloc_us(256 * 1024);
    // ping-pong recurrent state
    unsigned short *hHb[2] = { alloc_us(256 * 1024), alloc_us(256 * 1024) };
    unsigned short *hLb[2] = { alloc_us(256 * 1024), alloc_us(256 * 1024) };
    float *hFb[2] = { alloc_f(256 * 1024), alloc_f(256 * 1024) };
    float *sFb[2] = { alloc_f(256 * 64),   alloc_f(256 * 64) };

    // one-time (per call) weight conversion to swizzled bf16 hi/lo
    struct WDesc { const float* W; unsigned short *h, *l; int N, K; };
    WDesc wd[7] = {
        { Wsa, WsaH, WsaL, 1024, 96   },
        { Wih, WihH, WihL, 3072, 1024 },
        { Whh, WhhH, WhhL, 3072, 1024 },
        { Wp1, Wp1H, Wp1L, 1024, 1024 },
        { Wp2, Wp2H, Wp2L, 128,  1024 },
        { Wq1, Wq1H, Wq1L, 1024, 2048 },
        { Wq2, Wq2H, Wq2L, 128,  1024 },
    };
    for (int i = 0; i < 7; ++i) {
        int total = wd[i].N * wd[i].K;
        convert_weight_kernel<<<(total + 255) / 256, 256, 0, stream>>>(
            wd[i].W, wd[i].h, wd[i].l, wd[i].K, wd[i].K / 32, total);
    }

    init_state_kernel<<<(256 * 1024 + 255) / 256, 256, 0, stream>>>(
        hFb[0], hHb[0], hLb[0], sFb[0]);

    const int packTotal = 256 * 96 + 256 * 1024;
    for (int t = 0; t < 63; ++t) {
        const int cur = t & 1, nxt = cur ^ 1;
        float* outSlice = out + (size_t)t * 256 * 1408;

        // 1) pack [s|a] and obs_{t+1} to bf16 hi/lo
        pack_step_kernel<<<(packTotal + 255) / 256, 256, 0, stream>>>(
            sFb[cur], act, obs, t, saH, saL, obH, obL);

        // 2) e = ELU([s|a] @ Wsa^T + bsa)
        sa_embed_kernel<<<dim3(16, 8), 256, 0, stream>>>(
            saH, saL, WsaH, WsaL, bsa, eH, eL);

        // 3) GRU -> h_new
        gru_kernel<<<dim3(16, 8), 256, 0, stream>>>(
            eH, eL, hHb[cur], hLb[cur], hFb[cur],
            WihH, WihL, WhhH, WhhL, bih, bhh,
            hFb[nxt], hHb[nxt], hLb[nxt], outSlice);

        // 4) hidden heads: p1 and q1 in one launch
        heads_hidden_kernel<<<dim3(16, 8, 2), 256, 0, stream>>>(
            hHb[nxt], hLb[nxt], obH, obL,
            Wp1H, Wp1L, bp1, Wq1H, Wq1L, bq1,
            p1H, p1L, q1H, q1L);

        // 5) output heads: prior + posterior in one launch
        heads_out_kernel<<<dim3(16, 2), 128, 0, stream>>>(
            p1H, p1L, q1H, q1L,
            Wp2H, Wp2L, bp2, Wq2H, Wq2L, bq2,
            nzp + (size_t)t * 256 * 64, nzq + (size_t)t * 256 * 64,
            outSlice, sFb[nxt]);
    }
}
